// GraphTransformerLayer_36653250904480
// MI455X (gfx1250) — compile-verified, matrix operand order checked by reading, not run on hardware
//
#include <hip/hip_runtime.h>

typedef __attribute__((ext_vector_type(16))) _Float16 v16h;
typedef __attribute__((ext_vector_type(8)))  _Float16 v8h;
typedef __attribute__((ext_vector_type(4)))  _Float16 v4h;
typedef __attribute__((ext_vector_type(8)))  float    v8f;

#define D_MODEL 128
#define N_HEADS 4
#define DK      32
#define DIM_FF  256
#define BB      4
#define LL      2048
#define NSPLIT  4
#define NEGINF  (-1e9f)

// K-dim index held by (vgpr r, packed j, lane-half) for 16-bit A/B WMMA fragments
__device__ __forceinline__ int kmap(int r, int j, int half) {
  return ((r < 4) ? (2 * r) : (16 + 2 * (r - 4))) + 8 * half + j;
}

// Fragment element 2r+j == row[kmap(r,j,half)]; elements are two contiguous
// 8-value runs at row+8*half and row+16+8*half.
__device__ __forceinline__ v16h frag_from_f32(const float* __restrict__ row, int half) {
  const float4 c0 = *(const float4*)(row + 8 * half);
  const float4 c1 = *(const float4*)(row + 8 * half + 4);
  const float4 c2 = *(const float4*)(row + 8 * half + 16);
  const float4 c3 = *(const float4*)(row + 8 * half + 20);
  v16h f;
  f[0]  = (_Float16)c0.x; f[1]  = (_Float16)c0.y; f[2]  = (_Float16)c0.z; f[3]  = (_Float16)c0.w;
  f[4]  = (_Float16)c1.x; f[5]  = (_Float16)c1.y; f[6]  = (_Float16)c1.z; f[7]  = (_Float16)c1.w;
  f[8]  = (_Float16)c2.x; f[9]  = (_Float16)c2.y; f[10] = (_Float16)c2.z; f[11] = (_Float16)c2.w;
  f[12] = (_Float16)c3.x; f[13] = (_Float16)c3.y; f[14] = (_Float16)c3.z; f[15] = (_Float16)c3.w;
  return f;
}

// f16 source: the whole fragment is two 16-byte loads.
__device__ __forceinline__ v16h frag_from_f16(const _Float16* __restrict__ row, int half) {
  const v8h a = *(const v8h*)(row + 8 * half);
  const v8h b = *(const v8h*)(row + 16 + 8 * half);
  v16h f;
#pragma unroll
  for (int i = 0; i < 8; ++i) { f[i] = a[i]; f[8 + i] = b[i]; }
  return f;
}

// ---------------------------------------------------------------------------
// Generic C[M,N] = act( A[M,K] @ W[N,K]^T + bias[N] + residual[M,N] )
// One wave computes one 16x16 tile with v_wmma_f32_16x16x32_f16.
// ---------------------------------------------------------------------------
__global__ void gemm_wmma(const float* __restrict__ A, const float* __restrict__ W,
                          const float* __restrict__ bias, const float* __restrict__ residual,
                          float* __restrict__ C, int M, int N, int K, int relu) {
  const int wavesPerBlock = blockDim.x >> 5;
  const int tile = blockIdx.x * wavesPerBlock + (threadIdx.x >> 5);
  const int ntiles = N >> 4;
  const int mtiles = M >> 4;
  if (tile >= mtiles * ntiles) return;
  const int tm = tile / ntiles, tn = tile % ntiles;
  const int lane = threadIdx.x & 31;
  const int half = lane >> 4, li = lane & 15;

  v8f acc = {};
  const float* arow = A + (size_t)(tm * 16 + li) * K;
  const float* wrow = W + (size_t)(tn * 16 + li) * K;
  for (int kb = 0; kb < K; kb += 32) {
    const v16h af = frag_from_f32(arow + kb, half);
    const v16h bf = frag_from_f32(wrow + kb, half);
    acc = __builtin_amdgcn_wmma_f32_16x16x32_f16(false, af, false, bf,
                                                 (short)0, acc, false, false);
  }

  // Epilogue: one uniform residual branch, batched loads, 32-bit offsets.
  const int n = tn * 16 + li;
  const float bn = bias[n];
  const int base = (tm * 16 + 8 * half) * N + n;
  float res[8];
  if (residual) {
    const float* rp = residual + base;
#pragma unroll
    for (int r = 0; r < 8; ++r) res[r] = rp[r * N];
  } else {
#pragma unroll
    for (int r = 0; r < 8; ++r) res[r] = 0.0f;
  }
  float* cp = C + base;
#pragma unroll
  for (int r = 0; r < 8; ++r) {
    float v = acc[r] + bn + res[r];
    if (relu) v = fmaxf(v, 0.0f);
    cp[r * N] = v;
  }
}

// ---------------------------------------------------------------------------
// Pack q|k halves of qkv[B,L,384] (f32) into qk[B,L,256] (f16), same layout.
// ---------------------------------------------------------------------------
__global__ void pack_qk(const float* __restrict__ qkv, _Float16* __restrict__ qk) {
  const int e = (blockIdx.x * blockDim.x + threadIdx.x) * 4;
  const int row = e >> 8;
  const int col = e & 255;
  const float4 v = *(const float4*)(qkv + (size_t)row * 384 + col);
  v4h o;
  o[0] = (_Float16)v.x; o[1] = (_Float16)v.y; o[2] = (_Float16)v.z; o[3] = (_Float16)v.w;
  *(v4h*)(qk + (size_t)row * 256 + col) = o;
}

// ---------------------------------------------------------------------------
// Transpose V slice of qkv into vt[b,h,dk,L] (f16) via wave-private 32x33
// LDS tiles; coalesced loads and stores. One wave per 32(L) x 32(dk) tile.
// ---------------------------------------------------------------------------
__global__ void transpose_v(const float* __restrict__ qkv, _Float16* __restrict__ vt) {
  __shared__ float tile[4][32][33];
  const int warp = threadIdx.x >> 5;
  const int lane = threadIdx.x & 31;
  const int gw = blockIdx.x * 4 + warp;   // 0 .. B*H*(L/32)-1
  const int lt = gw & ((LL / 32) - 1);
  const int bh = gw >> 6;                 // b*4+h
  const int l0 = lt * 32;
  float (*T)[33] = tile[warp];
  const float* src = qkv + (size_t)((bh >> 2) * LL + l0) * 384 + 2 * D_MODEL + (bh & 3) * DK + lane;
#pragma unroll
  for (int i = 0; i < 32; ++i) T[i][lane] = src[(size_t)i * 384];
  asm volatile("s_wait_dscnt 0" ::: "memory");
  _Float16* dst = vt + (size_t)(bh * DK) * LL + l0 + lane;
#pragma unroll
  for (int i = 0; i < 32; ++i) dst[(size_t)i * LL] = (_Float16)T[lane][i];
}

// ---------------------------------------------------------------------------
// Flash-style attention, key-split 4 ways for memory-level parallelism.
// One wave per (b, head, 16-query tile, key-chunk of 512).
// Writes partial accumulators + (m, l) row stats; attn_combine merges them.
// ---------------------------------------------------------------------------
__global__ void attn_wmma(const _Float16* __restrict__ qk, const _Float16* __restrict__ vt,
                          const float* __restrict__ attn_bias,
                          const float* __restrict__ node_mask,
                          const unsigned char* __restrict__ sparse_mask,
                          float* __restrict__ part_acc, float* __restrict__ part_ml) {
  __shared__ alignas(16) _Float16 Pbuf[4][16 * 32];
  const int warp = threadIdx.x >> 5;
  const int gw = blockIdx.x * 4 + warp;        // 0 .. B*H*(L/16)*NSPLIT-1
  const int chunk = gw & (NSPLIT - 1);
  const int idx = gw >> 2;
  const int qt = idx & ((LL / 16) - 1);
  const int bh = idx >> 7;                     // b*4+h
  const int hh = bh & 3;
  const int bb = bh >> 2;
  const int q0 = qt * 16;
  const int k0 = chunk * (LL / NSPLIT);
  const int lane = threadIdx.x & 31;
  const int half = lane >> 4, li = lane & 15;
  const float SC = 0.17677669529663687f;       // 1/sqrt(32)
  _Float16* Pw = Pbuf[warp];

  const v16h qa = frag_from_f16(qk + (bb * LL + q0 + li) * 256 + hh * DK, half);
  const _Float16* kbase = qk + (bb * LL + li) * 256 + D_MODEL + hh * DK;
  const _Float16* vtb0  = vt + (bh * DK + li) * LL;        // dims 0..15
  const _Float16* vtb1  = vtb0 + 16 * LL;                  // dims 16..31
  const float* nmb = node_mask + bb * LL;

  float mrow[8], lsum[8];
#pragma unroll
  for (int r = 0; r < 8; ++r) { mrow[r] = -1e30f; lsum[r] = 0.0f; }
  v8f acc0 = {}, acc1 = {};

  for (int kb = k0; kb < k0 + LL / NSPLIT; kb += 32) {
    const v16h kf0 = frag_from_f16(kbase + kb * 256, half);
    const v16h kf1 = frag_from_f16(kbase + (kb + 16) * 256, half);
    v8f z = {};
    v8f s0 = __builtin_amdgcn_wmma_f32_16x16x32_f16(false, qa, false, kf0, (short)0, z, false, false);
    v8f s1 = __builtin_amdgcn_wmma_f32_16x16x32_f16(false, qa, false, kf1, (short)0, z, false, false);

    const float km0 = nmb[kb + li];
    const float km1 = nmb[kb + 16 + li];

#pragma unroll
    for (int r = 0; r < 8; ++r) {
      const int qi = q0 + r + 8 * half;
      const int brow = (bh * LL + qi) * LL + kb;
      const int srow = (bb * LL + qi) * LL + kb;
      const bool ok0 = sparse_mask[srow + li]      && (km0 > 0.0f);
      const bool ok1 = sparse_mask[srow + 16 + li] && (km1 > 0.0f);
      float x0 = ok0 ? (s0[r] * SC + attn_bias[brow + li])      : NEGINF;
      float x1 = ok1 ? (s1[r] * SC + attn_bias[brow + 16 + li]) : NEGINF;

      float t = fmaxf(x0, x1);
#pragma unroll
      for (int o = 8; o; o >>= 1) t = fmaxf(t, __shfl_xor(t, o, 16));
      const float nm = fmaxf(mrow[r], t);
      const float corr = __expf(mrow[r] - nm);
      const float p0 = __expf(x0 - nm);
      const float p1 = __expf(x1 - nm);
      float rs = p0 + p1;
#pragma unroll
      for (int o = 8; o; o >>= 1) rs += __shfl_xor(rs, o, 16);
      lsum[r] = lsum[r] * corr + rs;
      mrow[r] = nm;
      acc0[r] *= corr;
      acc1[r] *= corr;
      Pw[(r + 8 * half) * 32 + li] = (_Float16)p0;
      Pw[(r + 8 * half) * 32 + 16 + li] = (_Float16)p1;
    }
    asm volatile("s_wait_dscnt 0" ::: "memory");

    const v16h pa  = frag_from_f16(Pw + li * 32, half);
    const v16h vf0 = frag_from_f16(vtb0 + kb, half);
    const v16h vf1 = frag_from_f16(vtb1 + kb, half);
    acc0 = __builtin_amdgcn_wmma_f32_16x16x32_f16(false, pa, false, vf0, (short)0, acc0, false, false);
    acc1 = __builtin_amdgcn_wmma_f32_16x16x32_f16(false, pa, false, vf1, (short)0, acc1, false, false);
  }

  // --- write partials: acc[16][32], then m|l (replicated across half) ---
  float* pacc = part_acc + (size_t)gw * 512;
#pragma unroll
  for (int r = 0; r < 8; ++r) {
    const int row = r + 8 * half;
    pacc[row * 32 + li] = acc0[r];
    pacc[row * 32 + 16 + li] = acc1[r];
  }
  if (li == 0) {
    float* pml = part_ml + (size_t)gw * 32;
#pragma unroll
    for (int r = 0; r < 8; ++r) {
      pml[r + 8 * half] = mrow[r];
      pml[16 + r + 8 * half] = lsum[r];
    }
  }
}

// ---------------------------------------------------------------------------
// Merge NSPLIT partials per (b,h,16-query tile): exact online-softmax algebra.
// One wave per tile group; lane = head dim.
// ---------------------------------------------------------------------------
__global__ void attn_combine(const float* __restrict__ part_acc, const float* __restrict__ part_ml,
                             const float* __restrict__ node_mask, float* __restrict__ out) {
  const int warp = threadIdx.x >> 5;
  const int g = blockIdx.x * 4 + warp;         // 0 .. B*H*(L/16)-1
  const int qt = g & ((LL / 16) - 1);
  const int bh = g >> 7;
  const int hh = bh & 3;
  const int bb = bh >> 2;
  const int lane = threadIdx.x & 31;           // head dim
  const float* pa  = part_acc + (size_t)g * NSPLIT * 512;
  const float* pml = part_ml + (size_t)g * NSPLIT * 32;

  for (int row = 0; row < 16; ++row) {
    float M = -1e30f;
#pragma unroll
    for (int c = 0; c < NSPLIT; ++c) M = fmaxf(M, pml[c * 32 + row]);
    float l = 0.0f, val = 0.0f;
#pragma unroll
    for (int c = 0; c < NSPLIT; ++c) {
      const float w = __expf(pml[c * 32 + row] - M);
      l += pml[c * 32 + 16 + row] * w;
      val += pa[c * 512 + row * 32 + lane] * w;
    }
    const int m = qt * 16 + row;
    const float qm = (node_mask[bb * LL + m] > 0.0f) ? 1.0f : 0.0f;
    out[(size_t)(bb * LL + m) * D_MODEL + hh * DK + lane] = val * (qm / l);
  }
}

// ---------------------------------------------------------------------------
// LayerNorm over D=128; one wave per row, one float4 per lane.
// ---------------------------------------------------------------------------
__global__ void ln_kernel(const float* __restrict__ x, const float* __restrict__ g,
                          const float* __restrict__ b, const float* __restrict__ mask,
                          float* __restrict__ y) {
  const int row = blockIdx.x * (blockDim.x >> 5) + (threadIdx.x >> 5);
  const int lane = threadIdx.x & 31;
  const float4 v = *(const float4*)(x + (size_t)row * D_MODEL + lane * 4);
  float s = v.x + v.y + v.z + v.w;
#pragma unroll
  for (int o = 16; o; o >>= 1) s += __shfl_xor(s, o, 32);
  const float mu = s * (1.0f / D_MODEL);
  const float dx = v.x - mu, dy = v.y - mu, dz = v.z - mu, dw = v.w - mu;
  float vs = dx * dx + dy * dy + dz * dz + dw * dw;
#pragma unroll
  for (int o = 16; o; o >>= 1) vs += __shfl_xor(vs, o, 32);
  const float inv = rsqrtf(vs * (1.0f / D_MODEL) + 1e-5f);
  const float mk = mask ? ((mask[row] > 0.0f) ? 1.0f : 0.0f) : 1.0f;
  const float4 gg = *(const float4*)(g + lane * 4);
  const float4 bb4 = *(const float4*)(b + lane * 4);
  float4 o4;
  o4.x = (dx * inv * gg.x + bb4.x) * mk;
  o4.y = (dy * inv * gg.y + bb4.y) * mk;
  o4.z = (dz * inv * gg.z + bb4.z) * mk;
  o4.w = (dw * inv * gg.w + bb4.w) * mk;
  *(float4*)(y + (size_t)row * D_MODEL + lane * 4) = o4;
}

// ---------------------------------------------------------------------------
extern "C" void kernel_launch(void* const* d_in, const int* in_sizes, int n_in,
                              void* d_out, int out_size, void* d_ws, size_t ws_size,
                              hipStream_t stream) {
  const float* h      = (const float*)d_in[0];
  const float* bias   = (const float*)d_in[1];
  const float* nmask  = (const float*)d_in[2];
  const unsigned char* smask = (const unsigned char*)d_in[3];
  const float* w_qkv  = (const float*)d_in[4];
  const float* b_qkv  = (const float*)d_in[5];
  const float* w_out  = (const float*)d_in[6];
  const float* b_out  = (const float*)d_in[7];
  const float* w_ff1  = (const float*)d_in[8];
  const float* b_ff1  = (const float*)d_in[9];
  const float* w_ff2  = (const float*)d_in[10];
  const float* b_ff2  = (const float*)d_in[11];
  const float* ln1_g  = (const float*)d_in[12];
  const float* ln1_b  = (const float*)d_in[13];
  const float* ln2_g  = (const float*)d_in[14];
  const float* ln2_b  = (const float*)d_in[15];
  float* outp = (float*)d_out;

  const int rows = BB * LL;                 // 8192
  float* ws   = (float*)d_ws;
  float* qkv  = ws;                         // rows*384 f32
  float* att  = qkv + (size_t)rows * 384;   // rows*128 f32
  float* tmp  = att + (size_t)rows * 128;   // rows*128 f32
  float* h1   = tmp + (size_t)rows * 128;   // rows*128 f32
  float* ff   = h1  + (size_t)rows * 128;   // rows*256 f32
  _Float16* qk16 = (_Float16*)(ff + (size_t)rows * 256);   // rows*256 f16
  _Float16* vt16 = qk16 + (size_t)rows * 256;              // B*H*32*L f16
  float* part_ml = (float*)(vt16 + (size_t)BB * N_HEADS * DK * LL);  // 8192*32 f32
  // Partial accumulators (8192 waves * 512 f32 = 16 MB) alias tmp|h1|ff,
  // which are dead during attention and first consumed after attn_combine.
  float* part_acc = tmp;

  // 1) QKV projection
  gemm_wmma<<<(512 * 24) / 4, 128, 0, stream>>>(h, w_qkv, b_qkv, nullptr, qkv, rows, 384, 128, 0);
  // 2) pack q|k to f16; transpose V to vt[b,h,dk,L] f16
  pack_qk<<<(rows * 256 / 4) / 256, 256, 0, stream>>>(qkv, qk16);
  transpose_v<<<(BB * N_HEADS * (LL / 32)) / 4, 128, 0, stream>>>(qkv, vt16);
  // 3) key-split fused attention (one pass over attn_bias) + merge
  attn_wmma<<<(BB * N_HEADS * (LL / 16) * NSPLIT) / 4, 128, 0, stream>>>(
      qk16, vt16, bias, nmask, smask, part_acc, part_ml);
  attn_combine<<<(BB * N_HEADS * (LL / 16)) / 4, 128, 0, stream>>>(part_acc, part_ml, nmask, att);
  // 4) output projection + residual h   (overwrites part_acc region: OK)
  gemm_wmma<<<(512 * 8) / 4, 128, 0, stream>>>(att, w_out, b_out, h, tmp, rows, 128, 128, 0);
  // 5) LN1
  ln_kernel<<<rows / 8, 256, 0, stream>>>(tmp, ln1_g, ln1_b, nullptr, h1);
  // 6) FF1 + ReLU
  gemm_wmma<<<(512 * 16) / 4, 128, 0, stream>>>(h1, w_ff1, b_ff1, nullptr, ff, rows, 256, 128, 1);
  // 7) FF2 + residual h1
  gemm_wmma<<<(512 * 8) / 4, 128, 0, stream>>>(ff, w_ff2, b_ff2, h1, tmp, rows, 128, 256, 0);
  // 8) LN2 + node_mask
  ln_kernel<<<rows / 8, 256, 0, stream>>>(tmp, ln2_g, ln2_b, nmask, outp);
}
